// MonarchElmanCell_83236466196693
// MI455X (gfx1250) — compile-verified, matrix operand order checked by reading, not run on hardware
//
#include <hip/hip_runtime.h>
#include <hip/hip_bf16.h>

// ---------------------------------------------------------------------------
// MonarchElmanCell for MI455X (gfx1250, wave32, WMMA bf16 16x16x32)
//   D = 4096 = 64*64, T = 512, B = 16, rows = T*B = 8192
// ---------------------------------------------------------------------------

typedef __bf16 bf16_t;
typedef __attribute__((ext_vector_type(16))) __bf16 v16bf;
typedef __attribute__((ext_vector_type(8)))  float  v8f;

#define DIMD   4096
#define MBLK   64
#define ROWS   8192          // T*B
#define TSTEPS 512
#define BATCH  16

// ---- WMMA wrapper ----------------------------------------------------------
__device__ __forceinline__ v8f wmma_bf16(v16bf a, v16bf b, v8f c) {
  // (neg_a, A, neg_b, B, c_mod, C, reuse_a, reuse_b)
  return __builtin_amdgcn_wmma_f32_16x16x32_bf16(false, a, false, b,
                                                 (short)0, c, false, false);
}

// ---- fragment loader -------------------------------------------------------
// 16-bit A/B 16x32 per-lane layout (wave32):
//   lanes 0..15  : row/col = lane,    K = k0+{0..7} and k0+{16..23}
//   lanes 16..31 : row/col = lane-16, K = k0+{8..15} and k0+{24..31}
// Caller passes p = base + idx*ld + k0 + koff  (koff = (lane>=16)?8:0);
// fragment = 8 contiguous bf16 at p plus 8 contiguous bf16 at p+16.
__device__ __forceinline__ v16bf load_frag(const bf16_t* p) {
  v16bf f;
  __builtin_memcpy(&f, p, 16);
  __builtin_memcpy(((char*)&f) + 16, p + 16, 16);
  return f;
}

// ---- f32 -> bf16 bulk convert ---------------------------------------------
__global__ __launch_bounds__(256) void cvt_bf16_kernel(const float* __restrict__ src,
                                                       bf16_t* __restrict__ dst, int n) {
  int i = blockIdx.x * 256 + threadIdx.x;
  if (i < n) dst[i] = (bf16_t)src[i];
}

// ---------------------------------------------------------------------------
// Monarch over all rows (input branch), fused 2 stages per 16-row group.
// Stage 1: z[r,k,i] = sum_j B1[k,i,j]*x[r,k*64+j]; stored transposed to LDS
// as w[r, i*64+k].  Stage 2: Mx[r, a*64+i] = sum_j B2[a,i,j]*w[r, a*64+j].
// ---------------------------------------------------------------------------
__global__ __launch_bounds__(256) void monarch_x_kernel(const bf16_t* __restrict__ xh,
                                                        const bf16_t* __restrict__ B1,
                                                        const bf16_t* __restrict__ B2,
                                                        float* __restrict__ Mx) {
  __shared__ bf16_t w[BATCH * DIMD];              // 128 KB (<= 320 KB/WGP)
  const int row0 = blockIdx.x * 16;
  const int wave = threadIdx.x >> 5;
  const int lane = threadIdx.x & 31;
  const int half = lane >> 4;
  const int l    = lane & 15;
  const int koff = half * 8;

  for (int tile = wave; tile < 256; tile += 8) {
    const int k  = tile >> 2;
    const int nt = tile & 3;
    v8f acc = {};
#pragma unroll
    for (int kk = 0; kk < 64; kk += 32) {
      v16bf a = load_frag(xh + (size_t)(row0 + l) * DIMD + k * 64 + kk + koff);
      v16bf b = load_frag(B1 + ((k * 64) + nt * 16 + l) * 64 + kk + koff);
      acc = wmma_bf16(a, b, acc);
    }
    const int i = nt * 16 + l;                    // output column (i index)
#pragma unroll
    for (int v = 0; v < 8; ++v)
      w[(v + half * 8) * DIMD + i * 64 + k] = (bf16_t)acc[v];
  }
  __syncthreads();

  for (int tile = wave; tile < 256; tile += 8) {
    const int ab = tile >> 2;
    const int nt = tile & 3;
    v8f acc = {};
#pragma unroll
    for (int kk = 0; kk < 64; kk += 32) {
      v16bf a = load_frag(&w[l * DIMD + ab * 64 + kk + koff]);
      v16bf b = load_frag(B2 + ((ab * 64) + nt * 16 + l) * 64 + kk + koff);
      acc = wmma_bf16(a, b, acc);
    }
    const int col = ab * 64 + nt * 16 + l;
#pragma unroll
    for (int v = 0; v < 8; ++v)
      Mx[(size_t)(row0 + v + half * 8) * DIMD + col] = acc[v];
  }
}

// ---------------------------------------------------------------------------
// gate = silu(x @ W^T + b_gate).  W row-major [o][d] => both A and B are
// K-contiguous.  Wave tile 64x64 (4x4 frags), WG = 8 waves => 128x256 tile.
// ---------------------------------------------------------------------------
__global__ __launch_bounds__(256) void gate_kernel(const bf16_t* __restrict__ xh,
                                                   const bf16_t* __restrict__ Wh,
                                                   const float* __restrict__ bgate,
                                                   float* __restrict__ gate) {
  const int wave = threadIdx.x >> 5;
  const int lane = threadIdx.x & 31;
  const int half = lane >> 4;
  const int l    = lane & 15;
  const int koff = half * 8;
  const int wm = wave >> 2, wn = wave & 3;
  const int row0 = blockIdx.x * 128 + wm * 64;
  const int col0 = blockIdx.y * 256 + wn * 64;

  v8f acc[4][4];
#pragma unroll
  for (int i = 0; i < 4; ++i)
#pragma unroll
    for (int j = 0; j < 4; ++j) acc[i][j] = (v8f){};

  for (int k0 = 0; k0 < DIMD; k0 += 32) {
    v16bf af[4], bf[4];
#pragma unroll
    for (int i = 0; i < 4; ++i) {
      const bf16_t* pa = xh + (size_t)(row0 + i * 16 + l) * DIMD + k0 + koff;
      __builtin_prefetch(pa + 256, 0, 0);         // -> global_prefetch_b8
      af[i] = load_frag(pa);
    }
#pragma unroll
    for (int j = 0; j < 4; ++j) {
      const bf16_t* pb = Wh + (size_t)(col0 + j * 16 + l) * DIMD + k0 + koff;
      __builtin_prefetch(pb + 256, 0, 0);
      bf[j] = load_frag(pb);
    }
#pragma unroll
    for (int i = 0; i < 4; ++i)
#pragma unroll
      for (int j = 0; j < 4; ++j) acc[i][j] = wmma_bf16(af[i], bf[j], acc[i][j]);
  }

#pragma unroll
  for (int i = 0; i < 4; ++i)
#pragma unroll
    for (int j = 0; j < 4; ++j) {
      const int cc = col0 + j * 16 + l;
      const float bg = bgate[cc];
#pragma unroll
      for (int v = 0; v < 8; ++v) {
        const int rr = row0 + i * 16 + half * 8 + v;
        const float s = acc[i][j][v] + bg;
        gate[(size_t)rr * DIMD + cc] = s / (1.0f + __expf(-s));   // silu
      }
    }
}

// ---------------------------------------------------------------------------
// Sequential recurrence: single persistent WG (1024 threads = 32 waves).
//   h[t+1] = tanh(monarch(B1h,B2h,h[t]) + Mx[t] + b);  out[t] = h[t+1]*gate[t]
// h_prev lives in LDS as bf16 (128 KB) so the 512 dependent steps never
// round-trip state through global memory; the transpose buffer is another
// 128 KB (256 KB total, within the 320 KB WGP budget).
// ---------------------------------------------------------------------------
__global__ __launch_bounds__(1024) void recurrence_kernel(const float* __restrict__ h0,
                                                          const bf16_t* __restrict__ B1,
                                                          const bf16_t* __restrict__ B2,
                                                          const float* __restrict__ Mx,
                                                          const float* __restrict__ gate,
                                                          const float* __restrict__ bias,
                                                          float* __restrict__ out,
                                                          float* __restrict__ hout) {
  __shared__ bf16_t hs[BATCH * DIMD];             // 128 KB  (h state, bf16)
  __shared__ bf16_t w[BATCH * DIMD];              // 128 KB  (transpose buffer)
  const int tid  = threadIdx.x;
  const int wave = tid >> 5;
  const int lane = tid & 31;
  const int half = lane >> 4;
  const int l    = lane & 15;
  const int koff = half * 8;

  // h[0] = h0 (global, f32) and LDS state init
  for (int idx = tid; idx < BATCH * DIMD; idx += 1024) {
    const float v = h0[idx];
    hout[idx] = v;
    hs[idx]   = (bf16_t)v;
  }
  __syncthreads();

  for (int t = 0; t < TSTEPS; ++t) {
    // stage 1: z[r,k,i] from LDS h state; store transposed into w[r, i*64+k]
    for (int tile = wave; tile < 256; tile += 32) {
      const int k  = tile >> 2;
      const int nt = tile & 3;
      v8f acc = {};
#pragma unroll
      for (int kk = 0; kk < 64; kk += 32) {
        v16bf a = load_frag(&hs[l * DIMD + k * 64 + kk + koff]);
        v16bf b = load_frag(B1 + ((k * 64) + nt * 16 + l) * 64 + kk + koff);
        acc = wmma_bf16(a, b, acc);
      }
      const int i = nt * 16 + l;
#pragma unroll
      for (int v = 0; v < 8; ++v)
        w[(v + half * 8) * DIMD + i * 64 + k] = (bf16_t)acc[v];
    }
    __syncthreads();   // w complete; hs reads done -> epilogue may rewrite hs

    const float* mx_t = Mx   + (size_t)t * BATCH * DIMD;
    const float* g_t  = gate + (size_t)t * BATCH * DIMD;
    float* h_t = hout + (size_t)(t + 1) * BATCH * DIMD;
    float* o_t = out  + (size_t)t * BATCH * DIMD;

    // stage 2 + fused epilogue (tanh, h store, gated output, LDS state update)
    for (int tile = wave; tile < 256; tile += 32) {
      const int ab = tile >> 2;
      const int nt = tile & 3;
      v8f acc = {};
#pragma unroll
      for (int kk = 0; kk < 64; kk += 32) {
        v16bf a = load_frag(&w[l * DIMD + ab * 64 + kk + koff]);
        v16bf b = load_frag(B2 + ((ab * 64) + nt * 16 + l) * 64 + kk + koff);
        acc = wmma_bf16(a, b, acc);
      }
      const int col = ab * 64 + nt * 16 + l;
      // prefetch next timestep's streamed operands into cache
      __builtin_prefetch(mx_t + BATCH * DIMD + half * 8 * DIMD + col, 0, 0);
      __builtin_prefetch(g_t  + BATCH * DIMD + half * 8 * DIMD + col, 0, 0);
#pragma unroll
      for (int v = 0; v < 8; ++v) {
        const int r = v + half * 8;
        const float hn = tanhf(acc[v] + mx_t[r * DIMD + col] + bias[col]);
        h_t[r * DIMD + col] = hn;
        o_t[r * DIMD + col] = hn * g_t[r * DIMD + col];
        hs[r * DIMD + col]  = (bf16_t)hn;
      }
    }
    __syncthreads();   // hs update complete before next stage 1
  }
}

// ---------------------------------------------------------------------------
extern "C" void kernel_launch(void* const* d_in, const int* in_sizes, int n_in,
                              void* d_out, int out_size, void* d_ws, size_t ws_size,
                              hipStream_t stream) {
  (void)in_sizes; (void)n_in; (void)out_size; (void)ws_size;

  const float* x      = (const float*)d_in[0];   // [512,16,4096]
  const float* h0     = (const float*)d_in[1];   // [16,4096]
  const float* B1_h   = (const float*)d_in[2];   // [64,64,64]
  const float* B2_h   = (const float*)d_in[3];
  const float* B1_x   = (const float*)d_in[4];
  const float* B2_x   = (const float*)d_in[5];
  const float* W_gate = (const float*)d_in[6];   // [4096,4096]
  const float* b      = (const float*)d_in[7];
  const float* b_gate = (const float*)d_in[8];

  // workspace carve-up (all 256B aligned by construction)
  char* ws = (char*)d_ws;
  bf16_t* xh  = (bf16_t*)ws;  ws += (size_t)ROWS * DIMD * 2;        // 64 MB
  bf16_t* Wh  = (bf16_t*)ws;  ws += (size_t)DIMD * DIMD * 2;        // 32 MB
  bf16_t* b1x = (bf16_t*)ws;  ws += (size_t)MBLK * MBLK * MBLK * 2;
  bf16_t* b2x = (bf16_t*)ws;  ws += (size_t)MBLK * MBLK * MBLK * 2;
  bf16_t* b1h = (bf16_t*)ws;  ws += (size_t)MBLK * MBLK * MBLK * 2;
  bf16_t* b2h = (bf16_t*)ws;  ws += (size_t)MBLK * MBLK * MBLK * 2;
  float*  Mx  = (float*)ws;   ws += (size_t)ROWS * DIMD * 4;        // 134 MB
  float*  gt  = (float*)ws;   ws += (size_t)ROWS * DIMD * 4;        // 134 MB

  float* out_output = (float*)d_out;                       // [512,16,4096]
  float* out_h      = out_output + (size_t)ROWS * DIMD;    // [513,16,4096]

  const int NX = ROWS * DIMD;        // 33554432
  const int NW = DIMD * DIMD;        // 16777216
  const int NB = MBLK * MBLK * MBLK; // 262144

  cvt_bf16_kernel<<<(NX + 255) / 256, 256, 0, stream>>>(x, xh, NX);
  cvt_bf16_kernel<<<(NW + 255) / 256, 256, 0, stream>>>(W_gate, Wh, NW);
  cvt_bf16_kernel<<<(NB + 255) / 256, 256, 0, stream>>>(B1_x, b1x, NB);
  cvt_bf16_kernel<<<(NB + 255) / 256, 256, 0, stream>>>(B2_x, b2x, NB);
  cvt_bf16_kernel<<<(NB + 255) / 256, 256, 0, stream>>>(B1_h, b1h, NB);
  cvt_bf16_kernel<<<(NB + 255) / 256, 256, 0, stream>>>(B2_h, b2h, NB);

  monarch_x_kernel<<<ROWS / 16, 256, 0, stream>>>(xh, b1x, b2x, Mx);
  gate_kernel<<<dim3(ROWS / 128, DIMD / 256), 256, 0, stream>>>(xh, Wh, b_gate, gt);
  recurrence_kernel<<<1, 1024, 0, stream>>>(h0, b1h, b2h, Mx, gt, b,
                                            out_output, out_h);
}
